// CausalSelfAttention_20023137534917
// MI455X (gfx1250) — matrix operand order repaired and checked by reading, not compiled, not run on hardware
//
#include <hip/hip_runtime.h>

// ---------------------------------------------------------------------------
// CDNA5 (gfx1250) causal self-attention: bf16 WMMA GEMMs + flash attention.
// Wave32. Tensor Data Mover (TDM) async tile loads with double-buffered LDS,
// exp2-domain online softmax, diagonal-only causal masking, softmax
// denominator via WMMA (P x ones).
// ---------------------------------------------------------------------------

#define BB 4
#define TT 2048
#define CC 1024
#define NH 16
#define NKV 4
#define HD 64
#define NT (BB * TT) // 8192 rows

typedef __attribute__((ext_vector_type(16))) __bf16 v16bf;
typedef __attribute__((ext_vector_type(8)))  float  v8f;
typedef __attribute__((ext_vector_type(4)))  unsigned v4u;
typedef __attribute__((ext_vector_type(8)))  int v8i;
typedef __attribute__((ext_vector_type(4)))  int v4i;

#define DEVINL __device__ __forceinline__

DEVINL __bf16 f2bf(float f) {
  unsigned u = __builtin_bit_cast(unsigned, f);
  unsigned short h = (unsigned short)((u + 0x7FFFu + ((u >> 16) & 1u)) >> 16);
  return __builtin_bit_cast(__bf16, h);
}
DEVINL float bf2f(__bf16 x) {
  unsigned u = ((unsigned)__builtin_bit_cast(unsigned short, x)) << 16;
  return __builtin_bit_cast(float, u);
}

union V16U { v16bf v; uint4 q[2]; };

// A-fragment (16x32 bf16): lane L holds row L%16; lanes 0-15 hold K 0..7 &
// 16..23, lanes 16-31 hold K 8..15 & 24..31  (ISA 7.12.2 16-bit A layout).
DEVINL v16bf load_frag_a(const __bf16* tile, int rs) {
  int lane = threadIdx.x & 31;
  int r = lane & 15, sel = lane >> 4;
  V16U u;
  u.q[0] = *(const uint4*)(tile + r * rs + sel * 8);
  u.q[1] = *(const uint4*)(tile + r * rs + 16 + sel * 8);
  return u.v;
}

// B-fragment (32x16 bf16) from a TRANSPOSED tile (N rows x K cols): lane L
// holds column N=L%16; lanes 0-15 hold K 0..7 & 16..23, lanes 16-31 hold K 8..15 & 24..31.
DEVINL v16bf load_frag_b(const __bf16* bt, int rs) {
  int lane = threadIdx.x & 31;
  int n = lane & 15;
  const __bf16* p = bt + n * rs + ((lane >> 4) << 3);
  V16U u;
  u.q[0] = *(const uint4*)p;
  u.q[1] = *(const uint4*)(p + 16);
  return u.v;
}

DEVINL v8f wmma_bf16(v16bf a, v16bf b, v8f c) {
  return __builtin_amdgcn_wmma_f32_16x16x32_bf16(
      false, a, false, b, (short)0, c, false, false);
}

// ---------------------------------------------------------------------------
// TDM: async 2D tile load global -> LDS (ISA 08_async_tensor D# layout).
// data_size = 2 bytes; rows of tile0 (=64) elements; hardware pads the LDS
// destination with 4 DWORDs every 32 DWORDs -> 72-half padded rows.
//   tile0    : elements per row (x)
//   tile1    : number of rows (y)
//   stride0  : global row pitch in elements
// ---------------------------------------------------------------------------
DEVINL void tdm_load_2d(void* lds, const void* g, unsigned tile0,
                        unsigned tile1, unsigned long long stride0) {
  unsigned long long ga = (unsigned long long)(uintptr_t)g;
  v4u g0;
  g0.x = 1u;                               // count=1: one valid descriptor
  g0.y = (unsigned)(uintptr_t)lds;         // LDS byte address
  g0.z = (unsigned)ga;                     // global_addr[31:0]
  g0.w = (unsigned)(ga >> 32) | (2u << 30);  // global_addr[56:32] | type=2
  unsigned td0 = tile0, td1 = tile1;       // tensor dims = tile (exact fit)
  v8i g1;
  g1[0] = (int)((1u << 16)    // data_size = 2 bytes
              | (1u << 20)    // pad_enable
              | (4u << 22)    // pad_interval: 32 DWORDs
              | (3u << 25));  // pad_amount:   4 DWORDs
  g1[1] = (int)((td0 & 0xFFFFu) << 16);                 // tensor_dim0[15:0]
  g1[2] = (int)((td0 >> 16) | ((td1 & 0xFFFFu) << 16)); // td0 hi | td1 lo
  g1[3] = (int)((td1 >> 16) | (tile0 << 16));           // td1 hi | tile_dim0
  g1[4] = (int)tile1;                                   // tile_dim1 (dim2=0)
  g1[5] = (int)(unsigned)stride0;                       // dim0 stride lo32
  g1[6] = (int)(unsigned)(stride0 >> 32);               // dim0 stride hi16
  g1[7] = 0;
  v4i z4 = {0, 0, 0, 0};
#if __clang_major__ >= 23
  v8i z8 = {0, 0, 0, 0, 0, 0, 0, 0};
  __builtin_amdgcn_tensor_load_to_lds(g0, g1, z4, z4, z8, 0);
#else
  __builtin_amdgcn_tensor_load_to_lds(g0, g1, z4, z4, 0);
#endif
}

DEVINL void tdm_wait() { __builtin_amdgcn_s_wait_tensorcnt(0); }

// ---------------------------------------------------------------------------
// f32 -> bf16 conversion (plain and transposing variants)
// ---------------------------------------------------------------------------
__global__ __launch_bounds__(256) void cvt_f32_bf16(const float* __restrict__ in,
                                                    __bf16* __restrict__ out,
                                                    unsigned n) {
  unsigned i = blockIdx.x * 256 + threadIdx.x;
  if (i < n) out[i] = f2bf(in[i]);
}

// in: (K x N) row-major f32  ->  out: (N x K) row-major bf16
__global__ __launch_bounds__(256) void cvt_transpose(const float* __restrict__ in,
                                                     __bf16* __restrict__ out,
                                                     int K, int N) {
  unsigned i = blockIdx.x * 256 + threadIdx.x;
  if (i >= (unsigned)(K * N)) return;
  int k = i / (unsigned)N;
  int n = i - k * N;
  out[(size_t)n * K + k] = f2bf(in[i]);
}

// ---------------------------------------------------------------------------
// bf16 WMMA GEMM: C(MxN) = A(MxK) * Bt^T, with Bt pre-transposed (N x K).
// Block = 128 thr (4 waves), tile BM=128 x BN=64, BK=64.
// TDM double-buffered pipeline: tile i+1 DMAs while tile i is in the WMMAs.
// ---------------------------------------------------------------------------
template <bool F32OUT>
__global__ __launch_bounds__(128) void gemm_bf16(const __bf16* __restrict__ A,
                                                 const __bf16* __restrict__ Bt,
                                                 void* __restrict__ Cout,
                                                 int M, int N, int K) {
  constexpr int RSA = 72;  // 64 + 8 pad (halves) == TDM pad config
  constexpr int RSB = 72;
  __shared__ alignas(16) __bf16 lds_a[2][128 * RSA];
  __shared__ alignas(16) __bf16 lds_bt[2][64 * RSB];
  int tid = threadIdx.x;
  int wave = tid >> 5;
  int m0 = blockIdx.y * 128;
  int n0 = blockIdx.x * 64;

  v8f acc[2][4];
  #pragma unroll
  for (int ms = 0; ms < 2; ++ms)
    #pragma unroll
    for (int c = 0; c < 4; ++c) acc[ms][c] = v8f{};

  const __bf16* abase = A + (size_t)m0 * K;
  const __bf16* bbase = Bt + (size_t)n0 * K;

  if (wave == 0) {  // prime the pipeline: tile 0 via TDM
    tdm_load_2d(lds_a[0], abase, 64, 128, (unsigned long long)K);
    tdm_load_2d(lds_bt[0], bbase, 64, 64, (unsigned long long)K);
  }

  for (int k0 = 0; k0 < K; k0 += 64) {
    int cur = (k0 >> 6) & 1;
    tdm_wait();       // issuing wave drains TENSORcnt; no-op elsewhere
    __syncthreads();  // LDS tile visible to all waves
    if (k0 + 64 < K && wave == 0) {  // DMA next tile into the other buffer
      tdm_load_2d(lds_a[1 - cur], abase + (k0 + 64), 64, 128,
                  (unsigned long long)K);
      tdm_load_2d(lds_bt[1 - cur], bbase + (k0 + 64), 64, 64,
                  (unsigned long long)K);
    }
    #pragma unroll
    for (int kh = 0; kh < 2; ++kh) {
      v16bf af[2];
      #pragma unroll
      for (int ms = 0; ms < 2; ++ms)
        af[ms] =
            load_frag_a(lds_a[cur] + (wave * 32 + ms * 16) * RSA + kh * 32, RSA);
      #pragma unroll
      for (int c = 0; c < 4; ++c) {
        v16bf bf = load_frag_b(lds_bt[cur] + c * 16 * RSB + kh * 32, RSB);
        #pragma unroll
        for (int ms = 0; ms < 2; ++ms)
          acc[ms][c] = wmma_bf16(af[ms], bf, acc[ms][c]);
      }
    }
    __syncthreads();  // all reads of buffer `cur` done before it is refilled
  }

  int lane = tid & 31;
  int nl = lane & 15;
  int rhalf = (lane >> 4) << 3;
  #pragma unroll
  for (int ms = 0; ms < 2; ++ms) {
    int rbase = m0 + wave * 32 + ms * 16 + rhalf;
    #pragma unroll
    for (int c = 0; c < 4; ++c)
      #pragma unroll
      for (int r = 0; r < 8; ++r) {
        size_t idx = (size_t)(rbase + r) * N + n0 + c * 16 + nl;
        if (F32OUT)
          ((float*)Cout)[idx] = acc[ms][c][r];
        else
          ((__bf16*)Cout)[idx] = f2bf(acc[ms][c][r]);
      }
  }
}

// ---------------------------------------------------------------------------
// RoPE + relayout to head-major [b][h][t][d]; V relayout to [b][kh][d][t]
// ---------------------------------------------------------------------------
__global__ __launch_bounds__(256) void rope_q(const __bf16* __restrict__ Qb,
                                              __bf16* __restrict__ Qh) {
  unsigned i = blockIdx.x * 256 + threadIdx.x;  // B*NH*TT*32 = 2^22
  int j = i & 31;
  int t = (i >> 5) & (TT - 1);
  int h = (i >> 16) & (NH - 1);
  int b = i >> 20;
  size_t ib = ((size_t)(b * TT + t)) * (NH * HD) + h * HD;
  float x1 = bf2f(Qb[ib + 2 * j]);
  float x2 = bf2f(Qb[ib + 2 * j + 1]);
  float ang = (float)t * powf(10000.0f, -(float)(2 * j) / 64.0f);
  float cs = cosf(ang), sn = sinf(ang);
  size_t ob = (((size_t)b * NH + h) * TT + t) * HD + j;
  Qh[ob] = f2bf(x1 * cs - x2 * sn);
  Qh[ob + 32] = f2bf(x1 * sn + x2 * cs);
}

__global__ __launch_bounds__(256) void rope_k(const __bf16* __restrict__ Kb,
                                              __bf16* __restrict__ Kh) {
  unsigned i = blockIdx.x * 256 + threadIdx.x;  // B*NKV*TT*32 = 2^20
  int j = i & 31;
  int t = (i >> 5) & (TT - 1);
  int kh = (i >> 16) & (NKV - 1);
  int b = i >> 18;
  size_t ib = ((size_t)(b * TT + t)) * (NKV * HD) + kh * HD;
  float x1 = bf2f(Kb[ib + 2 * j]);
  float x2 = bf2f(Kb[ib + 2 * j + 1]);
  float ang = (float)t * powf(10000.0f, -(float)(2 * j) / 64.0f);
  float cs = cosf(ang), sn = sinf(ang);
  size_t ob = (((size_t)b * NKV + kh) * TT + t) * HD + j;
  Kh[ob] = f2bf(x1 * cs - x2 * sn);
  Kh[ob + 32] = f2bf(x1 * sn + x2 * cs);
}

// Vb (b,t,kh*64+d) -> Vt [b][kh][d][t]  (transposed for attention B-frags)
__global__ __launch_bounds__(256) void relayout_v(const __bf16* __restrict__ Vb,
                                                  __bf16* __restrict__ Vt) {
  unsigned i = blockIdx.x * 256 + threadIdx.x;  // B*NKV*HD*TT = 2^21
  int t = i & (TT - 1);
  int d = (i >> 11) & (HD - 1);
  int kh = (i >> 17) & (NKV - 1);
  int b = i >> 19;
  Vt[(((size_t)b * NKV + kh) * HD + d) * TT + t] =
      Vb[((size_t)(b * TT + t)) * (NKV * HD) + kh * HD + d];
}

// ---------------------------------------------------------------------------
// Flash attention: 128 thr (4 waves) per (b, h, 64 q-rows); wave owns 16 rows.
// 64 keys per iteration: 8 WMMA for S, 8 for P*V, 2 for l = P*ones.
// TDM double-buffered K/V tiles; exp2-domain softmax; diagonal-only mask.
// ---------------------------------------------------------------------------
__global__ __launch_bounds__(128) void attn_kernel(const __bf16* __restrict__ Qh,
                                                   const __bf16* __restrict__ Kh,
                                                   const __bf16* __restrict__ Vt,
                                                   __bf16* __restrict__ Y) {
  constexpr int RSK = 72;  // 64 + 8 pad == TDM pad config
  constexpr int RSV = 72;
  constexpr int RSP = 72;
  __shared__ alignas(16) __bf16 lds_k[2][64 * RSK];   // keys x d (Bt of K^T)
  __shared__ alignas(16) __bf16 lds_vt[2][HD * RSV];  // d x keys (Bt of V)
  __shared__ alignas(16) __bf16 lds_p[4][16 * RSP];   // per-wave P staging

  int tid = threadIdx.x, lane = tid & 31, wave = tid >> 5;
  int qt0 = blockIdx.x * 64;
  int h = blockIdx.y, b = blockIdx.z;
  int kh = h >> 2;  // GQA: 4 query heads per kv head

  const __bf16* qbase = Qh + ((size_t)b * NH + h) * TT * HD;
  const __bf16* kbase = Kh + ((size_t)b * NKV + kh) * TT * HD;
  const __bf16* vbase = Vt + ((size_t)b * NKV + kh) * HD * TT;

  // Q fragments for this wave's 16 rows (d 0..31 / 32..63), direct global.
  int r = lane & 15, sel = lane >> 4;
  const __bf16* qrow = qbase + (size_t)(qt0 + wave * 16 + r) * HD;
  V16U uq0, uq1;
  uq0.q[0] = *(const uint4*)(qrow + sel * 8);
  uq0.q[1] = *(const uint4*)(qrow + 16 + sel * 8);
  uq1.q[0] = *(const uint4*)(qrow + 32 + sel * 8);
  uq1.q[1] = *(const uint4*)(qrow + 48 + sel * 8);
  v16bf qf0 = uq0.v, qf1 = uq1.v;

  v16bf ones;
  #pragma unroll
  for (int i = 0; i < 16; ++i) ones[i] = f2bf(1.0f);

  v8f o[4] = {v8f{}, v8f{}, v8f{}, v8f{}};
  v8f lacc = v8f{};  // softmax denominator, same rescale as o
  float m[8];
  #pragma unroll
  for (int rr = 0; rr < 8; ++rr) m[rr] = -__builtin_inff();

  int rowhalf = (lane >> 4) << 3;
  int coln = lane & 15;

  int nblocks = blockIdx.x + 1;  // keys 0 .. qt0+63 in blocks of 64

  if (wave == 0) {  // prime: K tile (64 keys x 64 d), V^T tile (64 d x 64 keys)
    tdm_load_2d(lds_k[0], kbase, 64, 64, 64ull);
    tdm_load_2d(lds_vt[0], vbase, 64, 64, (unsigned long long)TT);
  }

  // exp2-domain scale: 1/sqrt(64) * log2(e)
  const float SCL = 0.125f * 1.4426950408889634f;

  for (int blk = 0; blk < nblocks; ++blk) {
    int s0 = blk * 64;
    int cur = blk & 1;
    tdm_wait();
    __syncthreads();
    if (blk + 1 < nblocks && wave == 0) {  // DMA next K/V tiles
      tdm_load_2d(lds_k[1 - cur], kbase + (size_t)(s0 + 64) * HD, 64, 64, 64ull);
      tdm_load_2d(lds_vt[1 - cur], vbase + (s0 + 64), 64, 64,
                  (unsigned long long)TT);
    }
    const bool diag = (blk == nblocks - 1);  // only block needing the mask

    // S = Q * K^T for four 16-key subtiles (d = 64 -> 2 WMMA each)
    v8f s[4];
    #pragma unroll
    for (int j = 0; j < 4; ++j) {
      v16bf kb0 = load_frag_b(lds_k[cur] + j * 16 * RSK, RSK);
      v16bf kb1 = load_frag_b(lds_k[cur] + j * 16 * RSK + 32, RSK);
      s[j] = wmma_bf16(qf0, kb0, v8f{});
      s[j] = wmma_bf16(qf1, kb1, s[j]);
    }

    // online softmax per row (row lives in one VGPR across a 16-lane group)
    #pragma unroll
    for (int rr = 0; rr < 8; ++rr) {
      float v0 = s[0][rr] * SCL;
      float v1 = s[1][rr] * SCL;
      float v2 = s[2][rr] * SCL;
      float v3 = s[3][rr] * SCL;
      if (diag) {  // uniform branch: causal mask only on diagonal block
        int row = qt0 + wave * 16 + rowhalf + rr;
        if (s0 + coln > row)      v0 = -__builtin_inff();
        if (s0 + 16 + coln > row) v1 = -__builtin_inff();
        if (s0 + 32 + coln > row) v2 = -__builtin_inff();
        if (s0 + 48 + coln > row) v3 = -__builtin_inff();
      }
      float mx = fmaxf(fmaxf(v0, v1), fmaxf(v2, v3));
      #pragma unroll
      for (int off = 8; off >= 1; off >>= 1)
        mx = fmaxf(mx, __shfl_xor(mx, off, 32));
      float mnew = fmaxf(m[rr], mx);
      float alpha = __builtin_amdgcn_exp2f(m[rr] - mnew);
      m[rr] = mnew;
      int prow = rowhalf + rr;
      lds_p[wave][prow * RSP + coln]      = f2bf(__builtin_amdgcn_exp2f(v0 - mnew));
      lds_p[wave][prow * RSP + 16 + coln] = f2bf(__builtin_amdgcn_exp2f(v1 - mnew));
      lds_p[wave][prow * RSP + 32 + coln] = f2bf(__builtin_amdgcn_exp2f(v2 - mnew));
      lds_p[wave][prow * RSP + 48 + coln] = f2bf(__builtin_amdgcn_exp2f(v3 - mnew));
      #pragma unroll
      for (int c = 0; c < 4; ++c) o[c][rr] *= alpha;
      lacc[rr] *= alpha;
    }
    // wave-local LDS RAW: wait for P stores before fragment reload
    asm volatile("s_wait_dscnt 0" ::: "memory");

    // O += P * V ;  lacc += P * ones (softmax denominator via matrix pipe)
    v16bf pf0 = load_frag_a(lds_p[wave], RSP);
    v16bf pf1 = load_frag_a(lds_p[wave] + 32, RSP);
    #pragma unroll
    for (int c = 0; c < 4; ++c) {
      v16bf vf0 = load_frag_b(lds_vt[cur] + c * 16 * RSV, RSV);
      v16bf vf1 = load_frag_b(lds_vt[cur] + c * 16 * RSV + 32, RSV);
      o[c] = wmma_bf16(pf0, vf0, o[c]);
      o[c] = wmma_bf16(pf1, vf1, o[c]);
    }
    lacc = wmma_bf16(pf0, ones, lacc);
    lacc = wmma_bf16(pf1, ones, lacc);
    __syncthreads();  // buffer `cur` fully consumed before it is refilled
  }

  // normalize and store y in [b][t][h*64+d] (row-major for the Wo GEMM)
  #pragma unroll
  for (int rr = 0; rr < 8; ++rr) {
    float inv = __builtin_amdgcn_rcpf(lacc[rr]);
    int t = qt0 + wave * 16 + rowhalf + rr;
    #pragma unroll
    for (int c = 0; c < 4; ++c) {
      int d = c * 16 + coln;
      Y[((size_t)b * TT + t) * (NH * HD) + h * HD + d] = f2bf(o[c][rr] * inv);
    }
  }
}

// ---------------------------------------------------------------------------
// Host launcher
// ---------------------------------------------------------------------------
extern "C" void kernel_launch(void* const* d_in, const int* in_sizes, int n_in,
                              void* d_out, int out_size, void* d_ws,
                              size_t ws_size, hipStream_t stream) {
  (void)in_sizes; (void)n_in; (void)out_size; (void)ws_size;
  const float* x = (const float*)d_in[0];
  const float* Wq = (const float*)d_in[1];
  const float* Wk = (const float*)d_in[2];
  const float* Wv = (const float*)d_in[3];
  const float* Wo = (const float*)d_in[4];

  char* ws = (char*)d_ws;
  size_t off = 0;
  auto alloc = [&](size_t bytes) {
    size_t o = off;
    off += (bytes + 255) & ~(size_t)255;
    return o;
  };
  __bf16* xb  = (__bf16*)(ws + alloc((size_t)NT * CC * 2));
  __bf16* wqt = (__bf16*)(ws + alloc((size_t)CC * NH * HD * 2));   // (N x K)
  __bf16* wkt = (__bf16*)(ws + alloc((size_t)CC * NKV * HD * 2));
  __bf16* wvt = (__bf16*)(ws + alloc((size_t)CC * NKV * HD * 2));
  __bf16* wot = (__bf16*)(ws + alloc((size_t)CC * CC * 2));
  __bf16* qb  = (__bf16*)(ws + alloc((size_t)NT * NH * HD * 2));
  __bf16* kb  = (__bf16*)(ws + alloc((size_t)NT * NKV * HD * 2));
  __bf16* vb  = (__bf16*)(ws + alloc((size_t)NT * NKV * HD * 2));
  __bf16* qh  = (__bf16*)(ws + alloc((size_t)NT * NH * HD * 2));
  __bf16* khh = (__bf16*)(ws + alloc((size_t)NT * NKV * HD * 2));
  __bf16* vt  = (__bf16*)(ws + alloc((size_t)NT * NKV * HD * 2));
  __bf16* yb  = (__bf16*)(ws + alloc((size_t)NT * NH * HD * 2));

  // 1) convert to bf16 (weights pre-transposed to N x K)
  {
    unsigned n;
    n = NT * CC;       cvt_f32_bf16<<<(n + 255) / 256, 256, 0, stream>>>(x, xb, n);
    n = CC * NH * HD;  cvt_transpose<<<(n + 255) / 256, 256, 0, stream>>>(Wq, wqt, CC, NH * HD);
    n = CC * NKV * HD; cvt_transpose<<<(n + 255) / 256, 256, 0, stream>>>(Wk, wkt, CC, NKV * HD);
    n = CC * NKV * HD; cvt_transpose<<<(n + 255) / 256, 256, 0, stream>>>(Wv, wvt, CC, NKV * HD);
    n = CC * CC;       cvt_transpose<<<(n + 255) / 256, 256, 0, stream>>>(Wo, wot, CC, CC);
  }

  // 2) QKV projections (bf16 out); BM=128, BN=64
  gemm_bf16<false><<<dim3((NH * HD) / 64, NT / 128), 128, 0, stream>>>(
      xb, wqt, qb, NT, NH * HD, CC);
  gemm_bf16<false><<<dim3((NKV * HD) / 64, NT / 128), 128, 0, stream>>>(
      xb, wkt, kb, NT, NKV * HD, CC);
  gemm_bf16<false><<<dim3((NKV * HD) / 64, NT / 128), 128, 0, stream>>>(
      xb, wvt, vb, NT, NKV * HD, CC);

  // 3) RoPE + head-major relayout; V transposed to [b][kh][d][t]
  rope_q<<<(BB * NH * TT * 32) / 256, 256, 0, stream>>>(qb, qh);
  rope_k<<<(BB * NKV * TT * 32) / 256, 256, 0, stream>>>(kb, khh);
  relayout_v<<<(BB * NKV * HD * TT) / 256, 256, 0, stream>>>(vb, vt);

  // 4) flash attention
  attn_kernel<<<dim3(TT / 64, NH, BB), 128, 0, stream>>>(qh, khh, vt, yb);

  // 5) output projection (f32 out)
  gemm_bf16<true><<<dim3(CC / 64, NT / 128), 128, 0, stream>>>(
      yb, wot, d_out, NT, CC, CC);
}
